// Decoder_19078244728976
// MI455X (gfx1250) — matrix operand order repaired and checked by reading, not compiled, not run on hardware
//
#include <hip/hip_runtime.h>
#include <hip/hip_bf16.h>

// ---------------------------------------------------------------------------
// LSTM(B=256,S=256,H=512) + FC(OUT=128) for MI455X (gfx1250, wave32, WMMA)
// Strategy: batch-parallel persistent blocks. 16 blocks x 16 batch rows.
//   - 512 threads/block = 16 wave32 -> 4 waves/SIMD for L2-latency hiding
//   - h tile (16x512) in LDS as bf16 (stride padded vs bank conflicts)
//   - c tile in f32 VGPRs; W_hh bf16 streamed from L2 (L2-resident, 192MB)
//   - per wave: 32-wide hidden slice, all 4 gates -> no cross-wave staging
//   - gate matmul: v_wmma_f32_16x16x32_bf16, 128 WMMAs / wave / step
//   - activations via raw v_exp_f32 / v_rcp_f32 (TRANS ops, co-execute)
//   - k-loop kept ROLLED (#pragma unroll 1): prevents the compiler from
//     hoisting the time-invariant weight fragments out of the time loop and
//     spilling them to scratch
// ---------------------------------------------------------------------------

#define BATCH 256
#define SEQ   256
#define HID   512
#define NOUT  128
#define BT    16          // batch rows per block
#define HS    520         // padded LDS stride (ushorts) for h tile
#define NTHR  512         // threads per block (16 wave32)

typedef __attribute__((ext_vector_type(16))) __bf16 v16bf;
typedef __attribute__((ext_vector_type(8)))  float  v8f;

struct Frag32B { uint4 lo; uint4 hi; };

__device__ __forceinline__ v16bf make_frag(uint4 lo, uint4 hi) {
    Frag32B f{lo, hi};
    return __builtin_bit_cast(v16bf, f);
}

__device__ __forceinline__ unsigned short f2bf(float x) {
    unsigned int u = __float_as_uint(x);
    u += 0x7FFFu + ((u >> 16) & 1u);   // round-to-nearest-even
    return (unsigned short)(u >> 16);
}

// sigmoid via v_exp_f32 + v_rcp_f32 (2 TRANS ops)
__device__ __forceinline__ float fast_sigm(float x) {
    return __builtin_amdgcn_rcpf(1.0f + __builtin_amdgcn_exp2f(-1.442695041f * x));
}
// tanh(x) = 1 - 2/(exp(2x)+1), saturates correctly for |x| large
__device__ __forceinline__ float fast_tanh(float x) {
    return 1.0f - 2.0f * __builtin_amdgcn_rcpf(1.0f + __builtin_amdgcn_exp2f(2.885390082f * x));
}

// A-fragment (16x32 bf16, M rows in LDS h tile, stride HS):
// lane<16:  M=lane,    K = kk+{0..7}  and kk+{16..23}
// lane>=16: M=lane-16, K = kk+{8..15} and kk+{24..31}
__device__ __forceinline__ v16bf load_a_lds(const unsigned short* h, int kk, int lane) {
    int m   = lane & 15;
    int off = (lane >= 16) ? 8 : 0;
    const unsigned short* p = h + m * HS + kk + off;
    uint4 lo = *(const uint4*)(p);
    uint4 hi = *(const uint4*)(p + 16);
    return make_frag(lo, hi);
}

// B-fragment (32x16 bf16). B column n == weight row n (contiguous in K):
// lane<16:  N=lane,    K = kk+{0..7}  and kk+{16..23}
// lane>=16: N=lane-16, K = kk+{8..15} and kk+{24..31}
__device__ __forceinline__ v16bf load_b_glb(const unsigned short* w, int n, int kk, int lane) {
    int off = (lane >= 16) ? 8 : 0;
    const unsigned short* p = w + (size_t)n * HID + kk + off;
    uint4 lo = *(const uint4*)(p);
    uint4 hi = *(const uint4*)(p + 16);
    return make_frag(lo, hi);
}

// ---------------------------------------------------------------------------
// Prep: f32 -> bf16 weight conversion + fused bias, one pass, grid-stride.
// ---------------------------------------------------------------------------
__global__ void lstm_prep_kernel(const float* __restrict__ Whh,
                                 const float* __restrict__ bih,
                                 const float* __restrict__ bhh,
                                 const float* __restrict__ Wfc,
                                 unsigned short* __restrict__ whh_b,
                                 unsigned short* __restrict__ wfc_b,
                                 float* __restrict__ bias_sum) {
    int tid    = blockIdx.x * blockDim.x + threadIdx.x;
    int stride = gridDim.x * blockDim.x;
    for (int i = tid; i < 4 * HID * HID; i += stride) whh_b[i] = f2bf(Whh[i]);
    for (int i = tid; i < NOUT * HID;   i += stride) wfc_b[i] = f2bf(Wfc[i]);
    for (int i = tid; i < 4 * HID;      i += stride) bias_sum[i] = bih[i] + bhh[i];
}

// ---------------------------------------------------------------------------
// Main persistent LSTM kernel: 16 blocks x 512 threads (16 wave32).
// Wave wv owns hidden slice [32*wv, 32*wv+32): 2 N-tiles per gate.
// ---------------------------------------------------------------------------
__launch_bounds__(NTHR, 1)
__global__ void lstm_fc_kernel(const float* __restrict__ y_hist,   // (B,S)
                               const float* __restrict__ h0,       // (B,H)
                               const float* __restrict__ c0,       // (B,H)
                               const float* __restrict__ Wih,      // (4H,)
                               const float* __restrict__ bfc,      // (OUT,)
                               const unsigned short* __restrict__ whh_b, // (4H,H) bf16
                               const unsigned short* __restrict__ wfc_b, // (OUT,H) bf16
                               const float* __restrict__ bias_sum, // (4H,)
                               float* __restrict__ out) {          // (B,OUT)
    __shared__ __align__(16) unsigned short h_sm[BT * HS];   // h tile, bf16
    __shared__ __align__(16) float y_sm[BT * SEQ];           // input tile
    __shared__ __align__(16) float bias_sm[4 * HID];
    __shared__ __align__(16) float wih_sm[4 * HID];

    const int tid  = threadIdx.x;
    const int lane = tid & 31;
    const int wv   = tid >> 5;            // wave id 0..15
    const int b0   = blockIdx.x * BT;     // batch base

    // ---- stage h0 (f32->bf16), y tile, bias, W_ih into LDS ----
    for (int i = tid; i < BT * HID; i += NTHR) {
        int m = i >> 9, k = i & (HID - 1);
        h_sm[m * HS + k] = f2bf(h0[(b0 + m) * HID + k]);
    }
    for (int i = tid; i < BT * SEQ; i += NTHR) {
        int m = i >> 8, s = i & (SEQ - 1);
        y_sm[m * SEQ + s] = y_hist[(b0 + m) * SEQ + s];
    }
    for (int i = tid; i < 4 * HID; i += NTHR) {
        bias_sm[i] = bias_sum[i];
        wih_sm[i]  = Wih[i];
    }
    __syncthreads();

    const int kbase = wv * 32;                 // this wave's hidden slice
    const int col   = lane & 15;               // N column within a tile
    const int mrow  = (lane >= 16) ? 8 : 0;    // C/D row offset for this lane

    // ---- cell state in registers: 2 tiles of 16x16 f32 (v8f each) ----
    v8f creg[2];
    #pragma unroll
    for (int tt = 0; tt < 2; ++tt)
        #pragma unroll
        for (int e = 0; e < 8; ++e)
            creg[tt][e] = c0[(size_t)(b0 + e + mrow) * HID + kbase + 16 * tt + col];

    // per-(gate,tile) scalars (constant over time for this lane)
    float wih_s[4][2], bias_s[4][2];
    #pragma unroll
    for (int g = 0; g < 4; ++g)
        #pragma unroll
        for (int tt = 0; tt < 2; ++tt) {
            int n = g * HID + kbase + 16 * tt + col;
            wih_s[g][tt]  = wih_sm[n];
            bias_s[g][tt] = bias_sm[n];
        }

    // ================= time loop (keep rolled!) =================
    #pragma unroll 1
    for (int st = 0; st < SEQ; ++st) {
        // x_t for this lane's 8 batch rows (independent of h -> load early)
        float xv[8];
        #pragma unroll
        for (int e = 0; e < 8; ++e) xv[e] = y_sm[(e + mrow) * SEQ + st];

        v8f acc[4][2];
        #pragma unroll
        for (int g = 0; g < 4; ++g)
            #pragma unroll
            for (int tt = 0; tt < 2; ++tt)
                acc[g][tt] = (v8f){0.f,0.f,0.f,0.f,0.f,0.f,0.f,0.f};

        // gates(16 x 128) += h(16 x 512) @ Whh_slice^T   via WMMA bf16.
        // KEEP ROLLED: a fully-unrolled k-loop lets the compiler hoist the
        // time-invariant weight fragments out of the st-loop and spill them
        // to scratch (disastrous). Rolled, the loads stream from L2 in
        // clauses and overlap with the WMMAs.
        #pragma unroll 1
        for (int kk = 0; kk < HID; kk += 32) {
            v16bf a = load_a_lds(h_sm, kk, lane);
            #pragma unroll
            for (int g = 0; g < 4; ++g) {
                #pragma unroll
                for (int tt = 0; tt < 2; ++tt) {
                    int n = g * HID + kbase + 16 * tt;   // weight-row base
                    v16bf b = load_b_glb(whh_b, n + col, kk, lane);
                    acc[g][tt] = __builtin_amdgcn_wmma_f32_16x16x32_bf16(
                        false, a, false, b, (short)0, acc[g][tt], false, false);
                }
            }
        }

        __syncthreads();   // all waves done reading h_sm for this step

        // ---- cell update + h write-back (bf16) ----
        #pragma unroll
        for (int tt = 0; tt < 2; ++tt) {
            int k = kbase + 16 * tt + col;
            #pragma unroll
            for (int e = 0; e < 8; ++e) {
                float iv = fast_sigm(acc[0][tt][e] + xv[e] * wih_s[0][tt] + bias_s[0][tt]);
                float fv = fast_sigm(acc[1][tt][e] + xv[e] * wih_s[1][tt] + bias_s[1][tt]);
                float gv = fast_tanh(acc[2][tt][e] + xv[e] * wih_s[2][tt] + bias_s[2][tt]);
                float ov = fast_sigm(acc[3][tt][e] + xv[e] * wih_s[3][tt] + bias_s[3][tt]);
                float cn = fv * creg[tt][e] + iv * gv;
                creg[tt][e] = cn;
                float hn = ov * fast_tanh(cn);
                h_sm[(e + mrow) * HS + k] = f2bf(hn);
            }
        }
        __syncthreads();   // h_new visible to everyone before next step
    }

    // ================= final FC: out = h @ Wfc^T + bfc =================
    // waves 0..7 handle output columns [16*wv, 16*wv+16)
    if (wv < 8) {
        v8f facc = (v8f){0.f,0.f,0.f,0.f,0.f,0.f,0.f,0.f};
        #pragma unroll 1
        for (int kk = 0; kk < HID; kk += 32) {
            v16bf a = load_a_lds(h_sm, kk, lane);
            v16bf b = load_b_glb(wfc_b, wv * 16 + col, kk, lane);
            facc = __builtin_amdgcn_wmma_f32_16x16x32_bf16(
                false, a, false, b, (short)0, facc, false, false);
        }
        float badd = bfc[wv * 16 + col];
        #pragma unroll
        for (int e = 0; e < 8; ++e)
            out[(size_t)(b0 + e + mrow) * NOUT + wv * 16 + col] = facc[e] + badd;
    }
}

// ---------------------------------------------------------------------------
extern "C" void kernel_launch(void* const* d_in, const int* in_sizes, int n_in,
                              void* d_out, int out_size, void* d_ws, size_t ws_size,
                              hipStream_t stream) {
    (void)in_sizes; (void)n_in; (void)out_size; (void)ws_size;

    const float* y_hist = (const float*)d_in[0];   // (B,S)
    const float* h0     = (const float*)d_in[1];   // (B,H)
    const float* c0     = (const float*)d_in[2];   // (B,H)
    const float* W_ih   = (const float*)d_in[3];   // (4H,1)
    const float* W_hh   = (const float*)d_in[4];   // (4H,H)
    const float* b_ih   = (const float*)d_in[5];   // (4H,)
    const float* b_hh   = (const float*)d_in[6];   // (4H,)
    const float* W_fc   = (const float*)d_in[7];   // (OUT,H)
    const float* b_fc   = (const float*)d_in[8];   // (OUT,)
    float* out = (float*)d_out;

    // workspace layout: [ Whh bf16 : 4H*H ][ Wfc bf16 : OUT*H ][ bias f32 : 4H ]
    unsigned short* whh_b = (unsigned short*)d_ws;
    unsigned short* wfc_b = whh_b + (size_t)4 * HID * HID;
    float* bias_sum = (float*)(wfc_b + (size_t)NOUT * HID);

    lstm_prep_kernel<<<2048, 256, 0, stream>>>(W_hh, b_ih, b_hh, W_fc,
                                               whh_b, wfc_b, bias_sum);

    lstm_fc_kernel<<<BATCH / BT, NTHR, 0, stream>>>(y_hist, h0, c0, W_ih, b_fc,
                                                    whh_b, wfc_b, bias_sum, out);
}
